// DICDNet_43198781063638
// MI455X (gfx1250) — matrix operand order repaired and checked by reading, not compiled, not run on hardware
//
#include <hip/hip_runtime.h>

typedef _Float16 half_t;
typedef __attribute__((ext_vector_type(16))) _Float16 v16h;
typedef __attribute__((ext_vector_type(8)))  float    v8f;

#define HW    256
#define NPIX  65536
#define NVIEW 360
#define ND    512
#define PI_F  3.14159265358979323846f
#define KSTAGE 512

// ---------------- ramp filter: h = irfft(ramp, 1024) (backward norm) ----------------
__global__ void k_ramp_h(const float* __restrict__ ramp, float* __restrict__ h) {
  int d = blockIdx.x * blockDim.x + threadIdx.x;
  if (d >= 1024) return;
  float acc = ramp[0] + ramp[512] * cosf(PI_F * (float)d);
  float w = 2.f * PI_F * (float)d / 1024.f;
  for (int k = 1; k < 512; ++k) acc += 2.f * ramp[k] * cosf(w * (float)k);
  h[d] = acc * (1.f / 1024.f);
}

// fs[v][m] = sum_n sino[v][n] * h[(m-n) mod 1024]   (circular conv; sino zero-padded)
__global__ void k_filter(const float* __restrict__ sino, const float* __restrict__ h,
                         float* __restrict__ fs) {
  int v = blockIdx.x;
  int m = blockIdx.y * blockDim.x + threadIdx.x;
  const float* row = sino + v * ND;
  float acc = 0.f;
  for (int n = 0; n < ND; ++n) acc += row[n] * h[(m - n) & 1023];
  fs[v * ND + m] = acc;
}

// fbp backprojection; writes fbp buffer and XQ channel 0
__global__ void k_backproj(const float* __restrict__ fs, const float* __restrict__ w2,
                           const float* __restrict__ frac, const int* __restrict__ didx,
                           half_t* __restrict__ fbp, half_t* __restrict__ xq0) {
  int p = blockIdx.x * blockDim.x + threadIdx.x;
  float acc = 0.f;
  for (int v = 0; v < NVIEW; ++v) {
    int q = v * NPIX + p;
    int i = didx[q];
    float f = frac[q];
    int i1 = (i + 1 > ND - 1) ? (ND - 1) : (i + 1);
    float g0 = fs[v * ND + i];
    float g1 = fs[v * ND + i1];
    acc += w2[v] * (g0 * (1.f - f) + g1 * f);
  }
  float val = acc * (PI_F / (float)NVIEW);
  fbp[p] = (half_t)val;
  xq0[p] = (half_t)val;
}

// ---------------- WMMA implicit-GEMM convolution ----------------
// Each wave: 16 (Cout) x 64 (pixels) output tile = 4 WMMA column tiles sharing one
// A (weight) fragment per 32-wide K chunk.
// out[co][p] = act( res[co][p] + ah*(*sp)*acc + bias[co] + cadd )
// wtrans: W(co,ci,ky,kx) = w[((ci*Cout+co)*R + (R-1-ky))*R + (R-1-kx)]  (conv-transpose)
__global__ __launch_bounds__(128) void k_conv_wmma(
    const half_t* __restrict__ in, const float* __restrict__ w,
    const float* __restrict__ bias, const half_t* __restrict__ res,
    half_t* __restrict__ out,
    int Cin, int Cout, int R, int wtrans, float wscale,
    float ah, const float* __restrict__ sp, float cadd, int do_relu)
{
  __shared__ half_t sW[KSTAGE * 16];   // [k][m] weight tile (f16)
  __shared__ int    sKD[KSTAGE];       // per-k decode: (ci<<16)|((dy+16)<<8)|(dx+16)

  const int RR   = R * R;
  const int Ktot = Cin * RR;
  const int Kpad = (Ktot + 31) & ~31;
  const int c    = R >> 1;

  const int lane = threadIdx.x & 31;
  const int wid  = threadIdx.x >> 5;
  const int tile = blockIdx.x * 4 + wid;     // 1024 tiles of 64 pixels
  const int y    = tile >> 2;                // row
  const int xb   = (tile & 3) << 6;          // 64-pixel column base
  const int cobase = blockIdx.y * 16;
  const int n    = lane & 15;
  const int hi   = lane >> 4;
  const int koff8  = hi * 8;
  const int koff16 = hi * 16;
  const float ahv = ah * (sp ? sp[0] : 1.f);

  // speculative prefetch of the rows this wave will touch (L2-resident anyway)
  {
    int yy = y + (lane & 7) - c;
    if (yy < 0) yy = 0; if (yy > HW - 1) yy = HW - 1;
    __builtin_prefetch(in + yy * HW + xb, 0, 3);
  }

  v8f acc0 = {}, acc1 = {}, acc2 = {}, acc3 = {};
  for (int kbase = 0; kbase < Kpad; kbase += KSTAGE) {
    int kcnt = Kpad - kbase; if (kcnt > KSTAGE) kcnt = KSTAGE;
    __syncthreads();
    // stage decode table
    for (int i2 = threadIdx.x; i2 < kcnt; i2 += 128) {
      int k = kbase + i2;
      int kd = (16 << 8) | 16; // k >= Ktot: ci=0, dy=dx=0 (weight is 0)
      if (k < Ktot) {
        int ci = k / RR; int t = k - ci * RR; int ky = t / R; int kx = t - ky * R;
        kd = (ci << 16) | ((ky - c + 16) << 8) | (kx - c + 16);
      }
      sKD[i2] = kd;
    }
    // stage weights as f16, layout [k][co-within-tile]
    for (int i2 = threadIdx.x; i2 < kcnt * 16; i2 += 128) {
      int kk = i2 >> 4; int mm = i2 & 15;
      int k = kbase + kk; int co = cobase + mm;
      float wv = 0.f;
      if (k < Ktot && co < Cout) {
        int ci = k / RR; int t = k - ci * RR; int ky = t / R; int kx = t - ky * R;
        int widx = wtrans ? (((ci * Cout + co) * R + (R - 1 - ky)) * R + (R - 1 - kx))
                          : (((co * Cin + ci) * R + ky) * R + kx);
        wv = w[widx] * wscale;
      }
      sW[i2] = (half_t)wv;
    }
    __syncthreads();

    for (int kc = 0; kc < kcnt; kc += 32) {
      v16h a, b0, b1, b2, b3;
      // A fragment: 16x32 f16, contiguous K (lanes<16: K 0-15; lanes>=16: K 16-31)
      #pragma unroll
      for (int j = 0; j < 16; ++j) {
        int ks = kc + koff16 + j;
        a[j] = sW[ks * 16 + n];
      }
      // B fragments for the 4 column subtiles; shared k-decode, per-subtile x check
      #pragma unroll
      for (int j = 0; j < 16; ++j) {
        int ks = kc + koff16 + j;
        int kd = sKD[ks];
        int ci = kd >> 16;
        int dy = ((kd >> 8) & 255) - 16;
        int dx = (kd & 255) - 16;
        int yy = y + dy;
        bool rowok = (unsigned)yy < 256u;
        const half_t* base = in + ci * NPIX + yy * HW;
        int x0 = xb + n + dx;
        half_t v0 = (half_t)0.f, v1 = (half_t)0.f, v2 = (half_t)0.f, v3 = (half_t)0.f;
        if (rowok) {
          if ((unsigned)(x0)      < 256u) v0 = base[x0];
          if ((unsigned)(x0 + 16) < 256u) v1 = base[x0 + 16];
          if ((unsigned)(x0 + 32) < 256u) v2 = base[x0 + 32];
          if ((unsigned)(x0 + 48) < 256u) v3 = base[x0 + 48];
        }
        b0[j] = v0; b1[j] = v1; b2[j] = v2; b3[j] = v3;
      }
      acc0 = __builtin_amdgcn_wmma_f32_16x16x32_f16(false, a, false, b0, (short)0, acc0, false, false);
      acc1 = __builtin_amdgcn_wmma_f32_16x16x32_f16(false, a, false, b1, (short)0, acc1, false, false);
      acc2 = __builtin_amdgcn_wmma_f32_16x16x32_f16(false, a, false, b2, (short)0, acc2, false, false);
      acc3 = __builtin_amdgcn_wmma_f32_16x16x32_f16(false, a, false, b3, (short)0, acc3, false, false);
    }
  }

  #pragma unroll
  for (int r = 0; r < 8; ++r) {
    int mm = hi * 8 + r;
    int co = cobase + mm;
    if (co >= Cout) continue;
    const size_t ob = (size_t)co * NPIX + y * HW + xb + n;
    float av[4] = {acc0[r], acc1[r], acc2[r], acc3[r]};
    #pragma unroll
    for (int s = 0; s < 4; ++s) {
      float v = ahv * av[s] + cadd;
      if (bias) v += bias[co];
      if (res)  v += (float)res[ob + s * 16];
      if (do_relu) v = fmaxf(v, 0.f);
      out[ob + s * 16] = (half_t)v;
    }
  }
}

// ---------------- elementwise glue ----------------
__global__ void k_fill1(half_t* __restrict__ x) {
  x[blockIdx.x * 256 + threadIdx.x] = (half_t)1.f;
}
__global__ void k_mask(half_t* __restrict__ g, const half_t* __restrict__ msk) {
  int i = blockIdx.x * 256 + threadIdx.x;
  if (!((float)msk[i] > 0.f)) g[i] = (half_t)0.f;
}
__global__ void k_relu_sub_tau(half_t* __restrict__ x, const float* __restrict__ tau) {
  int i = blockIdx.x * 256 + threadIdx.x;
  float v = (float)x[i] - tau[i >> 16];
  x[i] = (half_t)fmaxf(v, 0.f);
}
__global__ void k_ahat(half_t* __restrict__ t, const half_t* __restrict__ fbp,
                       const half_t* __restrict__ X, const half_t* __restrict__ FX,
                       const float* __restrict__ tau) {
  int i = blockIdx.x * 256 + threadIdx.x;
  float v = (float)fbp[i] - (float)X[i] + (float)FX[i] - tau[0];
  t[i] = (half_t)fmaxf(v, 0.f);
}
__global__ void k_resid(half_t* __restrict__ t, const half_t* __restrict__ A,
                        const half_t* __restrict__ fbp, const half_t* __restrict__ X,
                        const half_t* __restrict__ FX) {
  int i = blockIdx.x * 256 + threadIdx.x;
  t[i] = (half_t)((float)A[i] - (float)fbp[i] + (float)X[i] - (float)FX[i]);
}
__global__ void k_xmid(half_t* __restrict__ X, const half_t* __restrict__ fbp,
                       const half_t* __restrict__ A, const half_t* __restrict__ gm1,
                       const float* __restrict__ eta) {
  int i = blockIdx.x * 256 + threadIdx.x;
  float e = 0.1f * eta[0];
  float xv = (float)X[i];
  X[i] = (half_t)((1.f - e) * xv + e * ((float)fbp[i] - (float)A[i]) * (float)gm1[i]);
}
__global__ void k_out(float* __restrict__ o, const half_t* __restrict__ x) {
  int i = blockIdx.x * 256 + threadIdx.x;
  o[i] = (float)x[i];
}

// ---------------- host orchestration ----------------
extern "C" void kernel_launch(void* const* d_in, const int* in_sizes, int n_in,
                              void* d_out, int out_size, void* d_ws, size_t ws_size,
                              hipStream_t stream) {
  (void)in_sizes; (void)n_in; (void)out_size; (void)ws_size;
  const float* sino  = (const float*)d_in[0];
  const float* ramp  = (const float*)d_in[1];
  const float* w2    = (const float*)d_in[2];
  const float* dfrac = (const float*)d_in[3];
  const float* K_q   = (const float*)d_in[4];
  const float* K0    = (const float*)d_in[5];
  const float* K     = (const float*)d_in[6];
  const float* tau   = (const float*)d_in[7];
  const float* etaM  = (const float*)d_in[8];   // (9,1)
  const float* etaX  = (const float*)d_in[9];   // (10,1)
  const float* Xw    = (const float*)d_in[10];
  const float* Xb    = (const float*)d_in[11];
  const float* Mw    = (const float*)d_in[12];
  const float* Mb    = (const float*)d_in[13];
  const float* Mtau  = (const float*)d_in[14];
  const float* fw[6] = {(const float*)d_in[15], (const float*)d_in[17], (const float*)d_in[19],
                        (const float*)d_in[21], (const float*)d_in[23], (const float*)d_in[25]};
  const float* fb[6] = {(const float*)d_in[16], (const float*)d_in[18], (const float*)d_in[20],
                        (const float*)d_in[22], (const float*)d_in[24], (const float*)d_in[26]};
  const int* didx = (const int*)d_in[27];

  char* bump = (char*)d_ws;
  auto alloc = [&](size_t b) -> char* { char* r = bump; bump += (b + 255) & ~(size_t)255; return r; };
  float*  h   = (float*)alloc(1024 * 4);
  float*  fs  = (float*)alloc((size_t)NVIEW * ND * 4);
  half_t* fbp = (half_t*)alloc((size_t)NPIX * 2);
  half_t* XQ  = (half_t*)alloc((size_t)33 * NPIX * 2);
  half_t* T1  = (half_t*)alloc((size_t)33 * NPIX * 2);
  half_t* T2  = (half_t*)alloc((size_t)33 * NPIX * 2);
  half_t* M   = (half_t*)alloc((size_t)32 * NPIX * 2);
  half_t* F1  = (half_t*)alloc((size_t)16 * NPIX * 2);
  half_t* F2  = (half_t*)alloc((size_t)32 * NPIX * 2);
  half_t* F3  = (half_t*)alloc((size_t)32 * NPIX * 2);
  half_t* F4  = (half_t*)alloc((size_t)32 * NPIX * 2);
  half_t* F5  = (half_t*)alloc((size_t)16 * NPIX * 2);
  half_t* FXb = (half_t*)alloc((size_t)NPIX * 2);
  half_t* Ab  = (half_t*)alloc((size_t)NPIX * 2);
  half_t* gm1 = (half_t*)alloc((size_t)NPIX * 2);
  half_t* tin = (half_t*)alloc((size_t)NPIX * 2);

  auto conv = [&](const half_t* in, const float* w, const float* b, const half_t* res,
                  half_t* out, int Cin, int Cout, int R, int wtrans, float wscale,
                  float ah, const float* sp, float cadd, int relu) {
    dim3 g(256, (unsigned)((Cout + 15) / 16), 1);   // 256 blocks x 4 waves x 64 pixels
    k_conv_wmma<<<g, 128, 0, stream>>>(in, w, b, res, out, Cin, Cout, R, wtrans,
                                       wscale, ah, sp, cadd, relu);
  };
  auto xnet = [&](int i) {  // in-place on XQ (33 ch)
    for (int s = 0; s < 3; ++s) {
      size_t wo = (size_t)((i * 3 + s) * 2) * 33 * 33 * 9;
      size_t bo = (size_t)((i * 3 + s) * 2) * 33;
      conv(XQ, Xw + wo,             Xb + bo,      nullptr, T1, 33, 33, 3, 0, 1.f, 1.f, nullptr, 0.f, 1);
      conv(T1, Xw + wo + 33*33*9,   Xb + bo + 33, XQ,      XQ, 33, 33, 3, 0, 1.f, 1.f, nullptr, 0.f, s < 2 ? 1 : 0);
    }
  };
  auto mnet = [&](int i) {  // in-place on M (32 ch)
    for (int s = 0; s < 3; ++s) {
      size_t wo = (size_t)((i * 3 + s) * 2) * 32 * 32 * 9;
      size_t bo = (size_t)((i * 3 + s) * 2) * 32;
      conv(M,  Mw + wo,            Mb + bo,      nullptr, T1, 32, 32, 3, 0, 1.f, 1.f, nullptr, 0.f, 1);
      conv(T1, Mw + wo + 32*32*9,  Mb + bo + 32, M,       M,  32, 32, 3, 0, 1.f, 1.f, nullptr, 0.f, 1);
    }
    k_relu_sub_tau<<<32 * 256, 256, 0, stream>>>(M, Mtau + (size_t)i * 32);
  };
  auto fnet = [&]() {  // FX = f(X), X = XQ ch0; keeps ReLU activations for masks
    conv(XQ, fw[0], fb[0], nullptr, F1,  1, 16, 3, 0, 1.f, 1.f, nullptr, 0.f, 1);
    conv(F1, fw[1], fb[1], nullptr, F2, 16, 32, 3, 0, 1.f, 1.f, nullptr, 0.f, 0);
    conv(F2, fw[2], fb[2], nullptr, F3, 32, 32, 3, 0, 1.f, 1.f, nullptr, 0.f, 1);
    conv(F3, fw[3], fb[3], nullptr, F4, 32, 32, 3, 0, 1.f, 1.f, nullptr, 0.f, 0);
    conv(F4, fw[4], fb[4], nullptr, F5, 32, 16, 3, 0, 1.f, 1.f, nullptr, 0.f, 1);
    conv(F5, fw[5], fb[5], nullptr, FXb,16,  1, 3, 0, 1.f, 1.f, nullptr, 0.f, 0);
  };

  // ---- pipeline ----
  k_ramp_h <<<4, 256, 0, stream>>>(ramp, h);
  k_filter <<<dim3(NVIEW, 2), 256, 0, stream>>>(sino, h, fs);
  k_backproj<<<256, 256, 0, stream>>>(fs, w2, dfrac, didx, fbp, XQ);
  conv(fbp, K_q, nullptr, nullptr, XQ + NPIX, 1, 32, 3, 0, 1.f, 1.f, nullptr, 0.f, 0); // Q00
  xnet(0);                                   // XQ = [X0, Q0]
  fnet();                                    // FX0 + masks in F1/F3/F5

  // vjp(f, X0) applied to ones -> grad_X; gm1 = grad_X - 1
  k_fill1<<<256, 256, 0, stream>>>(Ab);
  conv(Ab, fw[5], nullptr, nullptr, T1,  1, 16, 3, 1, 1.f, 1.f, nullptr,  0.f, 0);
  k_mask<<<16 * 256, 256, 0, stream>>>(T1, F5);
  conv(T1, fw[4], nullptr, nullptr, T2, 16, 32, 3, 1, 1.f, 1.f, nullptr,  0.f, 0);
  conv(T2, fw[3], nullptr, nullptr, T1, 32, 32, 3, 1, 1.f, 1.f, nullptr,  0.f, 0);
  k_mask<<<32 * 256, 256, 0, stream>>>(T1, F3);
  conv(T1, fw[2], nullptr, nullptr, T2, 32, 32, 3, 1, 1.f, 1.f, nullptr,  0.f, 0);
  conv(T2, fw[1], nullptr, nullptr, T1, 32, 16, 3, 1, 1.f, 1.f, nullptr,  0.f, 0);
  k_mask<<<16 * 256, 256, 0, stream>>>(T1, F1);
  conv(T1, fw[0], nullptr, nullptr, gm1, 16, 1, 3, 1, 1.f, 1.f, nullptr, -1.f, 0);

  // Eps0 = convT(relu(fbp - X0 + FX0 - tau), K0/10) -> M; mnet; A = conv(M, K/10)
  k_ahat<<<256, 256, 0, stream>>>(tin, fbp, XQ, FXb, tau);
  conv(tin, K0, nullptr, nullptr, M, 1, 32, 9, 1, 0.1f, 1.f, nullptr, 0.f, 0);
  mnet(0);
  conv(M, K, nullptr, nullptr, Ab, 32, 1, 9, 0, 0.1f, 1.f, nullptr, 0.f, 0);
  k_xmid<<<256, 256, 0, stream>>>(XQ, fbp, Ab, gm1, etaX + 0);
  xnet(1);

  for (int k = 0; k < 9; ++k) {
    fnet();
    k_resid<<<256, 256, 0, stream>>>(tin, Ab, fbp, XQ, FXb);   // A - A_hat
    // M = M - (etaM/10) * convT(A - A_hat, K/10)
    conv(tin, K, nullptr, M, M, 1, 32, 9, 1, 0.1f, -0.1f, etaM + k, 0.f, 0);
    mnet(1 + k);
    conv(M, K, nullptr, nullptr, Ab, 32, 1, 9, 0, 0.1f, 1.f, nullptr, 0.f, 0);
    k_xmid<<<256, 256, 0, stream>>>(XQ, fbp, Ab, gm1, etaX + 1 + k);
    xnet(2 + k);
  }
  xnet(11);                                   // XQ_adjust
  k_out<<<256, 256, 0, stream>>>((float*)d_out, XQ);
}
